// MultiHeadAttention_17635135717395
// MI455X (gfx1250) — compile-verified, matrix operand order checked by reading, not run on hardware
//
#include <hip/hip_runtime.h>

typedef _Float16 half_t;
typedef __attribute__((ext_vector_type(16))) _Float16 v16h;
typedef __attribute__((ext_vector_type(8)))  float    v8f;
typedef unsigned int __attribute__((ext_vector_type(4))) v4u;
typedef int __attribute__((ext_vector_type(4)))  v4i;
typedef int __attribute__((ext_vector_type(8)))  v8i;

#define WMMA_F32_F16(a, b, c) \
  __builtin_amdgcn_wmma_f32_16x16x32_f16(false, (a), false, (b), (short)0, (c), false, false)

// K index inside a 16x32 f16 A/B fragment for half-wave g (0/1), element i (0..15).
// Per ISA: lanes 0-15 hold K={0..7,16..23}, lanes 16-31 hold K={8..15,24..31}.
__device__ __forceinline__ int frag_k(int g, int i) {
  int p = i >> 1, w = i & 1;
  return (p < 4) ? (g * 8 + p * 2 + w) : (16 + g * 8 + (p - 4) * 2 + w);
}

// TDM: load a dense 64x64 f16 tile (row stride 64 elems in memory) into LDS laid
// out as [64][72] f16 (16B pad per 128B row). D# packing per CDNA5 ISA ch.8:
//  g0: count=1 | lds_addr | global_addr(tile start) | type=2
//  g1: data_size=2B, pad_enable, pad_interval=4 (every 32 DW), pad_amount=3 (4 DW),
//      tensor_dim0=64, tensor_dim1=64, tile_dim0=64, tile_dim1=64, dim0_stride=64
__device__ __forceinline__ void tdm_load_tile64x64(const half_t* gtile, unsigned lds_off) {
  unsigned long long ga = (unsigned long long)gtile;
  v4u g0;
  g0.x = 1u;                                                  // count = 1 valid descriptor
  g0.y = lds_off;                                             // LDS byte address
  g0.z = (unsigned)(ga & 0xffffffffu);                        // global_addr[31:0]
  g0.w = (unsigned)((ga >> 32) & 0x01ffffffu) | 0x80000000u;  // addr[56:32] | type=2
  v8i g1;
  g1[0] = (1 << 16) | (1 << 20) | (4 << 22) | (3 << 25);      // 2B elems, pad 16B/128B
  g1[1] = (64 << 16);                                         // tensor_dim0[15:0]=64
  g1[2] = (64 << 16);                                         // td0 hi=0, tensor_dim1 lo=64
  g1[3] = (64 << 16);                                         // td1 hi=0, tile_dim0=64
  g1[4] = 64;                                                 // tile_dim1=64, tile_dim2=0
  g1[5] = 64;                                                 // tensor_dim0_stride=64
  g1[6] = 0;
  g1[7] = 0;
  v4i g2 = (v4i)0;
  v4i g3 = (v4i)0;
#if __clang_major__ >= 23
  v8i g4 = (v8i)0;
  __builtin_amdgcn_tensor_load_to_lds(g0, g1, g2, g3, g4, 0);
#else
  __builtin_amdgcn_tensor_load_to_lds(g0, g1, g2, g3, 0);
#endif
}

// ---------------------------------------------------------------- fp32 -> f16
__global__ void cvt_f32_f16(const float* __restrict__ src, half_t* __restrict__ dst, int n) {
  int i = blockIdx.x * blockDim.x + threadIdx.x;
  int stride = gridDim.x * blockDim.x;
  for (; i < n; i += stride) dst[i] = (half_t)src[i];
}

// ---------------------------------------------------------------- tiled GEMM
// C[M,N] = A[M,K] (f16, row-major) * W[K,N] (f16, row-major) + bias[N]
// MODE 0: scatter into Q (x0.125) / K / V buffers laid out [BH][S][DH] f16.
// MODE 1: fp32 output row-major.
// Block tile 128x128, K-step 32, 8 waves (2x4), each wave 64x32 (4x2 WMMA tiles).
template <int MODE>
__global__ __launch_bounds__(256) void gemm_f16(
    const half_t* __restrict__ A, const half_t* __restrict__ W,
    const float* __restrict__ bias, int M, int N, int K,
    half_t* __restrict__ qo, half_t* __restrict__ ko, half_t* __restrict__ vo,
    float* __restrict__ co) {
  __shared__ __align__(16) half_t As[128][40];   // [row][k]
  __shared__ __align__(16) half_t Bs[128][40];   // [col][k]  (transposed on store)

  const int tid  = threadIdx.x;
  const int lane = tid & 31;
  const int wave = tid >> 5;
  const int wm   = wave >> 2;          // 0..1
  const int wn   = wave & 3;           // 0..3
  const int rowBase = blockIdx.y * 128;
  const int colBase = blockIdx.x * 128;
  const int m16 = lane & 15, g = lane >> 4;

  v8f acc[4][2];
#pragma unroll
  for (int i = 0; i < 4; ++i)
#pragma unroll
    for (int j = 0; j < 2; ++j) acc[i][j] = (v8f)0.0f;

  for (int k0 = 0; k0 < K; k0 += 32) {
    // --- stage A tile 128x32
    {
      int r  = tid >> 2;
      int kc = (tid & 3) * 8;
#pragma unroll
      for (int rr = 0; rr < 2; ++rr) {
        int row = r + rr * 64;
        const half_t* src = A + (size_t)(rowBase + row) * K + k0 + kc;
        *(uint4*)&As[row][kc] = *(const uint4*)src;
      }
    }
    // --- stage B tile 32x128, transposed to [col][k]
    {
      int kk = tid >> 3;
      int nc = (tid & 7) * 16;
      const half_t* src = W + (size_t)(k0 + kk) * N + colBase + nc;
#pragma unroll
      for (int j = 0; j < 16; ++j) Bs[nc + j][kk] = src[j];
    }
    if (k0 + 32 < K) {  // speculative prefetch of next tiles
      __builtin_prefetch(A + (size_t)(rowBase + (tid >> 1)) * K + k0 + 32, 0, 1);
      __builtin_prefetch(W + (size_t)(k0 + 32 + (tid & 31)) * N + colBase, 0, 1);
    }
    __syncthreads();

    v16h af[4], bf[2];
#pragma unroll
    for (int ti = 0; ti < 4; ++ti) {
      int row = wm * 64 + ti * 16 + m16;
#pragma unroll
      for (int i = 0; i < 16; ++i) af[ti][i] = As[row][frag_k(g, i)];
    }
#pragma unroll
    for (int tj = 0; tj < 2; ++tj) {
      int col = wn * 32 + tj * 16 + m16;
#pragma unroll
      for (int i = 0; i < 16; ++i) bf[tj][i] = Bs[col][frag_k(g, i)];
    }
#pragma unroll
    for (int ti = 0; ti < 4; ++ti)
#pragma unroll
      for (int tj = 0; tj < 2; ++tj)
        acc[ti][tj] = WMMA_F32_F16(af[ti], bf[tj], acc[ti][tj]);
    __syncthreads();
  }

  // --- epilogue
#pragma unroll
  for (int ti = 0; ti < 4; ++ti)
#pragma unroll
    for (int tj = 0; tj < 2; ++tj)
#pragma unroll
      for (int r = 0; r < 8; ++r) {
        int row = rowBase + wm * 64 + ti * 16 + r + 8 * g;
        int col = colBase + wn * 32 + tj * 16 + m16;
        float v = acc[ti][tj][r] + bias[col];
        if constexpr (MODE == 0) {
          int b = row >> 11, sq = row & 2047;        // row = b*2048 + s
          int h = col / 192;
          int rem = col - h * 192;
          int d = rem / 3, t = rem - d * 3;          // col = h*192 + d*3 + t
          size_t idx = ((size_t)(b * 16 + h) * 2048 + sq) * 64 + d;
          if (t == 0)      qo[idx] = (half_t)(v * 0.125f);   // Q / sqrt(DH)
          else if (t == 1) ko[idx] = (half_t)v;
          else             vo[idx] = (half_t)v;
        } else {
          co[(size_t)row * N + col] = v;
        }
      }
}

// ---------------------------------------------------------------- flash attention
// Grid: (16 query-tiles, 64 heads). 8 waves; wave w owns queries [qb+16w, qb+16w+16).
// K/V staged into LDS by the Tensor Data Mover (wave 0 issues, TENSORcnt wait,
// then workgroup barrier). Online softmax in fp32.
__global__ __launch_bounds__(256) void flash_attn(
    const half_t* __restrict__ Qh, const half_t* __restrict__ Kh,
    const half_t* __restrict__ Vh, half_t* __restrict__ ctxh) {
  __shared__ __align__(16) half_t Ks[64][72];   // 128B row + 16B TDM pad
  __shared__ __align__(16) half_t Vs[64][72];
  __shared__ __align__(16) half_t Ps[8][16][72];

  const int tid = threadIdx.x, lane = tid & 31, wave = tid >> 5;
  const int bh = blockIdx.y;                 // 0..63
  const int qb = blockIdx.x * 128;
  const int b = bh >> 4, h = bh & 15;
  const int m16 = lane & 15, g = lane >> 4;
  const size_t headOff = (size_t)bh * 2048 * 64;
  const unsigned ldsK = (unsigned)(size_t)&Ks[0][0];   // low 32 bits = LDS offset
  const unsigned ldsV = (unsigned)(size_t)&Vs[0][0];

  // Q fragments (DH=64 -> two 32-wide K-steps), loaded straight from global.
  v16h qf[2];
  const int qrow = qb + wave * 16 + m16;     // A-fragment row for this lane
#pragma unroll
  for (int st = 0; st < 2; ++st)
#pragma unroll
    for (int i = 0; i < 16; ++i)
      qf[st][i] = Qh[headOff + (size_t)qrow * 64 + st * 32 + frag_k(g, i)];

  v8f acc[4];
#pragma unroll
  for (int t = 0; t < 4; ++t) acc[t] = (v8f)0.0f;
  float mrow[8], lrow[8];
#pragma unroll
  for (int r = 0; r < 8; ++r) { mrow[r] = -1e30f; lrow[r] = 0.f; }

  const int kblocks = (qb + 128) / 64;       // causal: keys <= qb+127
  const int wave_qmax = qb + wave * 16 + 15;

  for (int kb = 0; kb < kblocks; ++kb) {
    const int k0 = kb * 64;
    if (wave == 0) {  // TDM stage of K/V tiles (EXEC ignored; wave-uniform branch)
      tdm_load_tile64x64(Kh + headOff + (size_t)k0 * 64, ldsK);
      tdm_load_tile64x64(Vh + headOff + (size_t)k0 * 64, ldsV);
      __builtin_amdgcn_s_wait_tensorcnt(0);
    }
    __syncthreads();

    if (k0 <= wave_qmax) {  // wave-uniform -> EXEC stays all-ones for WMMA
      // scores S = Q @ K^T : 16 queries x 64 keys (4 C tiles)
      v8f sc[4];
#pragma unroll
      for (int t = 0; t < 4; ++t) sc[t] = (v8f)0.0f;
#pragma unroll
      for (int st = 0; st < 2; ++st)
#pragma unroll
        for (int tj = 0; tj < 4; ++tj) {
          v16h bf;  // B[d][key] = K[key][d]
#pragma unroll
          for (int i = 0; i < 16; ++i)
            bf[i] = Ks[tj * 16 + m16][st * 32 + frag_k(g, i)];
          sc[tj] = WMMA_F32_F16(qf[st], bf, sc[tj]);
        }

      // causal mask + per-row max
      float rmax[8];
#pragma unroll
      for (int r = 0; r < 8; ++r) rmax[r] = -1e30f;
#pragma unroll
      for (int tj = 0; tj < 4; ++tj) {
        int kg = k0 + tj * 16 + m16;
#pragma unroll
        for (int r = 0; r < 8; ++r) {
          int qg = qb + wave * 16 + r + 8 * g;
          float s = sc[tj][r];
          if (kg > qg) s = -1e30f;
          sc[tj][r] = s;
          rmax[r] = fmaxf(rmax[r], s);
        }
      }
#pragma unroll
      for (int r = 0; r < 8; ++r) {  // reduce across the 16-lane half-wave
        float v = rmax[r];
        v = fmaxf(v, __shfl_xor(v, 1, 32));
        v = fmaxf(v, __shfl_xor(v, 2, 32));
        v = fmaxf(v, __shfl_xor(v, 4, 32));
        v = fmaxf(v, __shfl_xor(v, 8, 32));
        rmax[r] = v;
      }

      float rscale[8], rsum[8];
#pragma unroll
      for (int r = 0; r < 8; ++r) {
        float mn = fmaxf(mrow[r], rmax[r]);
        rscale[r] = __expf(mrow[r] - mn);
        mrow[r] = mn;
        rsum[r] = 0.f;
      }

      // P = exp(S - m), spill to LDS to re-layout C->A fragment
#pragma unroll
      for (int tj = 0; tj < 4; ++tj) {
        int kg = k0 + tj * 16 + m16;
#pragma unroll
        for (int r = 0; r < 8; ++r) {
          int qg = qb + wave * 16 + r + 8 * g;
          float p = (kg > qg) ? 0.f : __expf(sc[tj][r] - mrow[r]);
          rsum[r] += p;
          Ps[wave][r + 8 * g][tj * 16 + m16] = (half_t)p;
        }
      }
#pragma unroll
      for (int r = 0; r < 8; ++r) {
        float v = rsum[r];
        v += __shfl_xor(v, 1, 32);
        v += __shfl_xor(v, 2, 32);
        v += __shfl_xor(v, 4, 32);
        v += __shfl_xor(v, 8, 32);
        lrow[r] = lrow[r] * rscale[r] + v;
      }
#pragma unroll
      for (int tj = 0; tj < 4; ++tj)
#pragma unroll
        for (int r = 0; r < 8; ++r) acc[tj][r] *= rscale[r];

      // ctx += P @ V   (LDS ops are in-order within a wave; same-wave RAW is safe)
#pragma unroll
      for (int st = 0; st < 2; ++st) {
        v16h pf;
#pragma unroll
        for (int i = 0; i < 16; ++i)
          pf[i] = Ps[wave][m16][st * 32 + frag_k(g, i)];
#pragma unroll
        for (int tj = 0; tj < 4; ++tj) {
          v16h vf;  // B[key][d]
#pragma unroll
          for (int i = 0; i < 16; ++i)
            vf[i] = Vs[st * 32 + frag_k(g, i)][tj * 16 + m16];
          acc[tj] = WMMA_F32_F16(pf, vf, acc[tj]);
        }
      }
    }
    __syncthreads();
  }

  // write ctx/l to [B,S,H*DH] f16 (input layout of the out-projection GEMM)
#pragma unroll
  for (int tj = 0; tj < 4; ++tj)
#pragma unroll
    for (int r = 0; r < 8; ++r) {
      int qg = qb + wave * 16 + r + 8 * g;
      int d = tj * 16 + m16;
      float v = acc[tj][r] / lrow[r];
      ctxh[((size_t)(b * 2048 + qg)) * 1024 + h * 64 + d] = (half_t)v;
    }
}

// ---------------------------------------------------------------- launch
extern "C" void kernel_launch(void* const* d_in, const int* in_sizes, int n_in,
                              void* d_out, int out_size, void* d_ws, size_t ws_size,
                              hipStream_t stream) {
  const float* query = (const float*)d_in[0];
  // d_in[1]: mask (causal) — computed analytically, not read
  const float* W_qkv = (const float*)d_in[2];
  const float* b_qkv = (const float*)d_in[3];
  const float* W_out = (const float*)d_in[4];
  const float* b_out = (const float*)d_in[5];

  char* ws = (char*)d_ws;
  size_t off = 0;
  auto alloc = [&](size_t bytes) -> void* {
    void* p = ws + off;
    off += (bytes + 255) & ~(size_t)255;
    return p;
  };
  const size_t TOK = 8192;  // B*S
  half_t* Xh  = (half_t*)alloc(TOK * 1024 * 2);
  half_t* Wqh = (half_t*)alloc((size_t)1024 * 3072 * 2);
  half_t* Woh = (half_t*)alloc((size_t)1024 * 1024 * 2);
  half_t* Qh  = (half_t*)alloc((size_t)64 * 2048 * 64 * 2);
  half_t* Kh  = (half_t*)alloc((size_t)64 * 2048 * 64 * 2);
  half_t* Vh  = (half_t*)alloc((size_t)64 * 2048 * 64 * 2);
  half_t* Ch  = (half_t*)alloc(TOK * 1024 * 2);

  cvt_f32_f16<<<4096, 256, 0, stream>>>(query, Xh, (int)(TOK * 1024));
  cvt_f32_f16<<<4096, 256, 0, stream>>>(W_qkv, Wqh, 1024 * 3072);
  cvt_f32_f16<<<2048, 256, 0, stream>>>(W_out, Woh, 1024 * 1024);

  gemm_f16<0><<<dim3(3072 / 128, 8192 / 128), 256, 0, stream>>>(
      Xh, Wqh, b_qkv, 8192, 3072, 1024, Qh, Kh, Vh, nullptr);

  flash_attn<<<dim3(16, 64), 256, 0, stream>>>(Qh, Kh, Vh, Ch);

  gemm_f16<1><<<dim3(1024 / 128, 8192 / 128), 256, 0, stream>>>(
      Ch, Woh, b_out, 8192, 1024, 1024, nullptr, nullptr, nullptr, (float*)d_out);
}
